// BF15IntLinear_31739808317850
// MI455X (gfx1250) — matrix operand order repaired and checked by reading, not compiled, not run on hardware
//
#include <hip/hip_runtime.h>
#include <hip/hip_bf16.h>

typedef __attribute__((ext_vector_type(16))) __bf16 bf16x16;
typedef __attribute__((ext_vector_type(8)))  float  f32x8;
typedef __attribute__((ext_vector_type(4)))  unsigned int u32x4;
typedef __attribute__((ext_vector_type(8)))  int  i32x8;
typedef __attribute__((ext_vector_type(4)))  int  i32x4;

#define TILE_M 128
#define TILE_N 256
#define TILE_K 32
#define LDSW   40          // 32 data + 4-DWORD TDM pad per row -> 80B row stride
#define K_DIM  1024
#define N_DIM  4096

__device__ __forceinline__ unsigned short bf15_trunc(float f) {
  unsigned u = __builtin_bit_cast(unsigned, f);
  return (unsigned short)((u >> 16) & 0xFFFCu);
}
__device__ __forceinline__ unsigned short bf16_rne(float f) {
  unsigned u = __builtin_bit_cast(unsigned, f);
  unsigned r = u + 0x7FFFu + ((u >> 16) & 1u);
  return (unsigned short)(r >> 16);
}

struct Q2 { uint4 lo, hi; };

// ---------------- pre-conversion kernels (bandwidth bound) ----------------
__global__ void cvt_x_bf15(const float4* __restrict__ in, ushort4* __restrict__ out, long n4) {
  long i = (long)blockIdx.x * blockDim.x + threadIdx.x;
  const long stride = (long)gridDim.x * blockDim.x;
  for (; i < n4; i += stride) {
    float4 v = in[i];
    ushort4 o;
    o.x = bf15_trunc(v.x); o.y = bf15_trunc(v.y);
    o.z = bf15_trunc(v.z); o.w = bf15_trunc(v.w);
    out[i] = o;
  }
}
__global__ void cvt_w_bf16(const float4* __restrict__ in, ushort4* __restrict__ out, long n4) {
  long i = (long)blockIdx.x * blockDim.x + threadIdx.x;
  const long stride = (long)gridDim.x * blockDim.x;
  for (; i < n4; i += stride) {
    float4 v = in[i];
    ushort4 o;
    o.x = bf16_rne(v.x); o.y = bf16_rne(v.y);
    o.z = bf16_rne(v.z); o.w = bf16_rne(v.w);
    out[i] = o;
  }
}

// ---------------- TDM helpers ----------------
// D# group1 for a 2D bf16 tile [tileRows x 32] out of a [tensorRows x 1024] row-major tensor.
// data_size=1 (2B), pad_enable, pad_interval=3 (16 DWORDs = one 64B row), pad_amount=3 (4 DWORDs = 16B)
__device__ __forceinline__ i32x8 tdm_group1(int tileRows, int tensorRows) {
  i32x8 g;
  g[0] = (1 << 16) | (1 << 20) | (3 << 22) | (3 << 25);
  g[1] = (int)((K_DIM & 0xFFFF) << 16);                              // tensor_dim0 lo16
  g[2] = (int)(((K_DIM >> 16) & 0xFFFF) | ((tensorRows & 0xFFFF) << 16)); // dim0 hi | dim1 lo
  g[3] = (int)(((tensorRows >> 16) & 0xFFFF) | (TILE_K << 16));      // dim1 hi | tile_dim0
  g[4] = tileRows;                                                   // tile_dim1 | tile_dim2=0
  g[5] = K_DIM;                                                      // tensor_dim0_stride lo32
  g[6] = 0;
  g[7] = 0;
  return g;
}
__device__ __forceinline__ void tdm_load_2d(unsigned lds_byte, unsigned long long gaddr, i32x8 g1) {
  u32x4 g0;
  g0.x = 1u;                                         // count=1, user descriptor
  g0.y = lds_byte;                                   // LDS byte address
  g0.z = (unsigned)gaddr;                            // global_addr[31:0]
  g0.w = (unsigned)((gaddr >> 32) & 0x01FFFFFFull) | 0x80000000u; // addr[56:32] | type=2
  i32x4 z4 = {0, 0, 0, 0};
  i32x8 z8 = {0, 0, 0, 0, 0, 0, 0, 0};
  // amdgpu-toolchain (clang-23) 6-arg form: (g0, g1, g2, g3, extra, cpol)
  __builtin_amdgcn_tensor_load_to_lds(g0, g1, z4, z4, z8, 0);
}

// ---------------- main GEMM: bf16 in ws, TDM -> LDS -> WMMA ----------------
__global__ __launch_bounds__(256)
void bf15_linear_wmma_tdm(const unsigned short* __restrict__ Xb,
                          const unsigned short* __restrict__ Wb,
                          const float* __restrict__ bias,
                          __hip_bfloat16* __restrict__ Y) {
  __shared__ unsigned short lds[2 * TILE_M * LDSW + 2 * TILE_N * LDSW]; // 60 KB

  const int tid  = threadIdx.x;
  const int lane = tid & 31;
  const int w    = tid >> 5;
  const int wm   = w & 1;        // 0..1 : 64-row M slab
  const int wn   = w >> 1;       // 0..3 : 64-col N slab
  const int half = lane >> 4;
  const int lr   = lane & 15;

  const long blockM = (long)blockIdx.y * TILE_M;
  const long blockN = (long)blockIdx.x * TILE_N;

  const int AOF0 = 0;
  const int AOF1 = TILE_M * LDSW;
  const int BOF0 = 2 * TILE_M * LDSW;
  const int BOF1 = 2 * TILE_M * LDSW + TILE_N * LDSW;

  f32x8 acc[4][4];
#pragma unroll
  for (int mi = 0; mi < 4; ++mi)
#pragma unroll
    for (int ni = 0; ni < 4; ++ni)
      acc[mi][ni] = (f32x8){0.f,0.f,0.f,0.f,0.f,0.f,0.f,0.f};

  const i32x8 g1A = tdm_group1(TILE_M, (int)(gridDim.y * TILE_M)); // M rows
  const i32x8 g1B = tdm_group1(TILE_N, N_DIM);

  // ISA 10.2: for LDS, generic address low 32 bits == LDS byte offset
  const unsigned ldsA[2] = { (unsigned)((uintptr_t)&lds[AOF0]),
                             (unsigned)((uintptr_t)&lds[AOF1]) };
  const unsigned ldsB[2] = { (unsigned)((uintptr_t)&lds[BOF0]),
                             (unsigned)((uintptr_t)&lds[BOF1]) };
  const unsigned long long gA0 = (unsigned long long)(uintptr_t)(Xb + blockM * K_DIM);
  const unsigned long long gB0 = (unsigned long long)(uintptr_t)(Wb + blockN * K_DIM);

  auto tdm_stage = [&](int t, int s) {
    const unsigned long long koff = (unsigned long long)t * TILE_K * 2ull; // bytes
    tdm_load_2d(ldsA[s], gA0 + koff, g1A);
    tdm_load_2d(ldsB[s], gB0 + koff, g1B);
  };

  auto compute = [&](int s) {
    const unsigned short* As = &lds[s ? AOF1 : AOF0];
    const unsigned short* Bsp = &lds[s ? BOF1 : BOF0];
    bf16x16 af[4], bfr[4];
#pragma unroll
    for (int mi = 0; mi < 4; ++mi) {
      const int r = (wm * 64 + mi * 16 + lr) * LDSW;
      Q2 q;
      q.lo = *(const uint4*)&As[r + half * 8];
      q.hi = *(const uint4*)&As[r + 16 + half * 8];
      af[mi] = __builtin_bit_cast(bf16x16, q);
    }
#pragma unroll
    for (int ni = 0; ni < 4; ++ni) {
      const int r = (wn * 64 + ni * 16 + lr) * LDSW;
      Q2 q;
      q.lo = *(const uint4*)&Bsp[r + half * 8];
      q.hi = *(const uint4*)&Bsp[r + 16 + half * 8];
      bfr[ni] = __builtin_bit_cast(bf16x16, q);
    }
#pragma unroll
    for (int mi = 0; mi < 4; ++mi)
#pragma unroll
      for (int ni = 0; ni < 4; ++ni)
        acc[mi][ni] = __builtin_amdgcn_wmma_f32_16x16x32_bf16(
            false, af[mi], false, bfr[ni], (short)0, acc[mi][ni], false, false);
  };

  if (w == 0) tdm_stage(0, 0);
  const int T = K_DIM / TILE_K;   // 32
  for (int t = 0; t < T; ++t) {
    if (t + 1 < T && w == 0) tdm_stage(t + 1, (t + 1) & 1);
    if (t + 1 < T) __builtin_amdgcn_s_wait_tensorcnt(2);  // stage t done, t+1 in flight
    else           __builtin_amdgcn_s_wait_tensorcnt(0);
    __syncthreads();               // all waves see stage t data
    compute(t & 1);
    __syncthreads();               // reads done before TDM overwrites this buffer
  }

#pragma unroll
  for (int ni = 0; ni < 4; ++ni) {
    const long gn = blockN + wn * 64 + ni * 16 + lr;
    const float bv = bias[gn];
#pragma unroll
    for (int mi = 0; mi < 4; ++mi) {
      const long gm0 = blockM + wm * 64 + mi * 16 + half * 8;
#pragma unroll
      for (int v = 0; v < 8; ++v) {
        __hip_bfloat16 h = __float2bfloat16(acc[mi][ni][v]);
        float y2 = __bfloat162float(h) + bv;
        Y[(gm0 + v) * (long)N_DIM + gn] = __float2bfloat16(y2);
      }
    }
  }
}

// ---------------- fallback: fused fp32->bf16 GEMM (round-2 kernel) ----------------
__global__ __launch_bounds__(256)
void bf15_linear_wmma_fused(const float* __restrict__ X,
                            const float* __restrict__ W,
                            const float* __restrict__ bias,
                            __hip_bfloat16* __restrict__ Y) {
  __shared__ unsigned short As[2][TILE_M * LDSW];
  __shared__ unsigned short Bs[2][TILE_N * LDSW];

  const int tid  = threadIdx.x;
  const int lane = tid & 31;
  const int w    = tid >> 5;
  const int wm   = w & 1;
  const int wn   = w >> 1;
  const int half = lane >> 4;
  const int lr   = lane & 15;

  const long blockM = (long)blockIdx.y * TILE_M;
  const long blockN = (long)blockIdx.x * TILE_N;

  const int colf4 = tid & 7;
  const int row0  = tid >> 3;
  const float* aPtr = X + (blockM + row0) * (long)K_DIM + colf4 * 4;
  const float* bPtr = W + (blockN + row0) * (long)K_DIM + colf4 * 4;

  float4 aR[4], bR[8];

  auto g_load = [&](int t) {
    const float* ap = aPtr + t * TILE_K;
    const float* bp = bPtr + t * TILE_K;
#pragma unroll
    for (int i = 0; i < 4; ++i)
      aR[i] = *(const float4*)(ap + (long)i * 32 * K_DIM);
#pragma unroll
    for (int i = 0; i < 8; ++i)
      bR[i] = *(const float4*)(bp + (long)i * 32 * K_DIM);
  };

  auto lds_store = [&](int s) {
#pragma unroll
    for (int i = 0; i < 4; ++i) {
      const int r = row0 + i * 32;
      ushort4 pa;
      pa.x = bf15_trunc(aR[i].x); pa.y = bf15_trunc(aR[i].y);
      pa.z = bf15_trunc(aR[i].z); pa.w = bf15_trunc(aR[i].w);
      *(ushort4*)&As[s][r * LDSW + colf4 * 4] = pa;
    }
#pragma unroll
    for (int i = 0; i < 8; ++i) {
      const int r = row0 + i * 32;
      ushort4 pb;
      pb.x = bf16_rne(bR[i].x); pb.y = bf16_rne(bR[i].y);
      pb.z = bf16_rne(bR[i].z); pb.w = bf16_rne(bR[i].w);
      *(ushort4*)&Bs[s][r * LDSW + colf4 * 4] = pb;
    }
  };

  f32x8 acc[4][4];
#pragma unroll
  for (int mi = 0; mi < 4; ++mi)
#pragma unroll
    for (int ni = 0; ni < 4; ++ni)
      acc[mi][ni] = (f32x8){0.f,0.f,0.f,0.f,0.f,0.f,0.f,0.f};

  auto compute = [&](int s) {
    bf16x16 af[4], bfr[4];
#pragma unroll
    for (int mi = 0; mi < 4; ++mi) {
      const int r = (wm * 64 + mi * 16 + lr) * LDSW;
      Q2 q;
      q.lo = *(const uint4*)&As[s][r + half * 8];
      q.hi = *(const uint4*)&As[s][r + 16 + half * 8];
      af[mi] = __builtin_bit_cast(bf16x16, q);
    }
#pragma unroll
    for (int ni = 0; ni < 4; ++ni) {
      const int r = (wn * 64 + ni * 16 + lr) * LDSW;
      Q2 q;
      q.lo = *(const uint4*)&Bs[s][r + half * 8];
      q.hi = *(const uint4*)&Bs[s][r + 16 + half * 8];
      bfr[ni] = __builtin_bit_cast(bf16x16, q);
    }
#pragma unroll
    for (int mi = 0; mi < 4; ++mi)
#pragma unroll
      for (int ni = 0; ni < 4; ++ni)
        acc[mi][ni] = __builtin_amdgcn_wmma_f32_16x16x32_bf16(
            false, af[mi], false, bfr[ni], (short)0, acc[mi][ni], false, false);
  };

  g_load(0);
  lds_store(0);
  __syncthreads();
  const int T = K_DIM / TILE_K;
  for (int t = 0; t < T; ++t) {
    if (t + 1 < T) g_load(t + 1);
    compute(t & 1);
    __syncthreads();
    if (t + 1 < T) {
      lds_store((t + 1) & 1);
      __syncthreads();
    }
  }

#pragma unroll
  for (int ni = 0; ni < 4; ++ni) {
    const long gn = blockN + wn * 64 + ni * 16 + lr;
    const float bv = bias[gn];
#pragma unroll
    for (int mi = 0; mi < 4; ++mi) {
      const long gm0 = blockM + wm * 64 + mi * 16 + half * 8;
#pragma unroll
      for (int v = 0; v < 8; ++v) {
        __hip_bfloat16 h = __float2bfloat16(acc[mi][ni][v]);
        float y2 = __bfloat162float(h) + bv;
        Y[(gm0 + v) * (long)N_DIM + gn] = __float2bfloat16(y2);
      }
    }
  }
}

extern "C" void kernel_launch(void* const* d_in, const int* in_sizes, int n_in,
                              void* d_out, int out_size, void* d_ws, size_t ws_size,
                              hipStream_t stream) {
  const float* x = (const float*)d_in[0];   // [B,S,1024] fp32
  const float* w = (const float*)d_in[1];   // [4096,1024] fp32
  const float* b = (const float*)d_in[2];   // [4096] fp32
  __hip_bfloat16* y = (__hip_bfloat16*)d_out;

  const long nX = in_sizes[0];                    // 33,554,432
  const long nW = in_sizes[1];                    // 4,194,304
  const long M  = nX / K_DIM;                     // 32768
  dim3 grid(N_DIM / TILE_N, (unsigned)(M / TILE_M));

  const size_t need = (size_t)nX * 2 + (size_t)nW * 2;   // 72 MB of bf16 staging
  if (ws_size >= need) {
    unsigned short* xb = (unsigned short*)d_ws;
    unsigned short* wb = xb + nX;
    cvt_x_bf15<<<dim3(4096), dim3(256), 0, stream>>>((const float4*)x, (ushort4*)xb, nX / 4);
    cvt_w_bf16<<<dim3(1024), dim3(256), 0, stream>>>((const float4*)w, (ushort4*)wb, nW / 4);
    bf15_linear_wmma_tdm<<<grid, dim3(256), 0, stream>>>(xb, wb, b, y);
  } else {
    bf15_linear_wmma_fused<<<grid, dim3(256), 0, stream>>>(x, w, b, y);
  }
}
